// gaussian_decoder_het2_58566174049039
// MI455X (gfx1250) — matrix operand order repaired and checked by reading, not compiled, not run on hardware
//
#include <hip/hip_runtime.h>
#include <hip/hip_bf16.h>
#include <math.h>

typedef __attribute__((ext_vector_type(16))) _Float16 v16h;
typedef __attribute__((ext_vector_type(8)))  float    v8f;

#define BOXN   256
#define KSZ    9
#define NPTS   20000
#define NBATCH 16
#define NNEU   32
#define NLAY   3
#define LATD   8

#define OUT_IMG_OFF 0
#define OUT_PD_OFF  (NBATCH * BOXN * BOXN)                 // 1,048,576
#define OUT_RES_OFF (OUT_PD_OFF + NBATCH * NPTS * 3)       // 2,008,576
#define OUT_AC_OFF  (OUT_RES_OFF + NBATCH * NPTS * 3)      // 2,968,576

// Weight tiles (f16, wave32 WMMA B-operand layout), one N-half (32K x 16N) each:
//  0..1  : linamp1_W  (K=4 padded to 32)        halves 0,1
//  2..7  : ampblock_W layers 0..2               (2 halves each)
//  8     : linamp2_W  (N=1 padded to 16)
//  9..10 : lin0_W     (K=10 padded to 32)       halves 0,1
// 11..16 : deform_W layers 0..2                 (2 halves each)
// 17     : lin1a_W    (N=3 padded to 16)
#define NTILES 18

__device__ inline v8f wmma16(v16h a, v16h b, v8f c) {
  // v_wmma_f32_16x16x32_f16: (neg_a, A, neg_b, B, c_mod, C, reuse_a, reuse_b)
  return __builtin_amdgcn_wmma_f32_16x16x32_f16(false, a, false, b, (short)0, c,
                                                false, false);
}

// A operand: 16x32 f16. lane<16 : M=lane, lane>=16 : M=lane-16 (K shifted by 8)
// VGPR v (pair j=2v,2v+1): K = (v>=4?16:0) + (lane>=16?8:0) + 2*(v&3) + {0,1}
__device__ inline v16h load_A16(const float* act, int lane) {
  union { v16h v; _Float16 e[16]; } a;
  const int m  = lane & 15;
  const int kb = (lane >= 16) ? 8 : 0;
#pragma unroll
  for (int v = 0; v < 8; ++v) {
    const int k = ((v >= 4) ? 16 : 0) + kb + ((v & 3) << 1);
    a.e[2 * v]     = (_Float16)act[m * 32 + k];
    a.e[2 * v + 1] = (_Float16)act[m * 32 + k + 1];
  }
  return a.v;
}

// B operand tile: lane reads 16 contiguous halves (32B, aligned) from LDS.
__device__ inline v16h load_B16(const _Float16* wT, int t, int lane) {
  return *(const v16h*)(wT + t * 512 + lane * 16);
}

// C/D layout: VGPR v <-> row M = v + (lane>=16?8:0), col N = half*16 + lane%16
__device__ inline void store_C(float* act, int lane, v8f c, int half) {
  const int n  = half * 16 + (lane & 15);
  const int rb = (lane >= 16) ? 8 : 0;
#pragma unroll
  for (int v = 0; v < 8; ++v) act[(rb + v) * 32 + n] = c[v];
}

// One head: embed GEMM (no bias), NLAY residual relu layers, output GEMM.
// act must hold the zero-padded [16 x 32] f32 input rows. Returns final D.
__device__ inline v8f run_head(float* act, const _Float16* wT, int lane,
                               int tIn, int tLay, const float* bias, int tOut) {
  v8f z = {};
  v16h A = load_A16(act, lane);
  v8f c0 = wmma16(A, load_B16(wT, tIn,     lane), z);
  v8f c1 = wmma16(A, load_B16(wT, tIn + 1, lane), z);
#pragma unroll
  for (int i = 0; i < NLAY; ++i) {
    store_C(act, lane, c0, 0);
    store_C(act, lane, c1, 1);
    v16h Ai = load_A16(act, lane);
    const float b0 = bias[i * NNEU + (lane & 15)];
    const float b1 = bias[i * NNEU + 16 + (lane & 15)];
    v8f cb0, cb1;
#pragma unroll
    for (int j = 0; j < 8; ++j) { cb0[j] = b0; cb1[j] = b1; }
    v8f d0 = wmma16(Ai, load_B16(wT, tLay + 2 * i,     lane), cb0);
    v8f d1 = wmma16(Ai, load_B16(wT, tLay + 2 * i + 1, lane), cb1);
#pragma unroll
    for (int j = 0; j < 8; ++j) {            // x = relu(x@W^T + b) + x
      c0[j] = fmaxf(d0[j], 0.f) + c0[j];
      c1[j] = fmaxf(d1[j], 0.f) + c1[j];
    }
  }
  store_C(act, lane, c0, 0);
  store_C(act, lane, c1, 1);
  v16h Af = load_A16(act, lane);
  return wmma16(Af, load_B16(wT, tOut, lane), z);
}

__global__ __launch_bounds__(256) void zero_img_kernel(float* img) {
  img[blockIdx.x * 256 + threadIdx.x] = 0.f;
}

__global__ __launch_bounds__(256)
void decoder_kernel(const float* __restrict__ z, const float* __restrict__ r,
                    const float* __restrict__ pos, const float* __restrict__ amp,
                    const float* __restrict__ linamp1_W,
                    const float* __restrict__ ampblock_W,
                    const float* __restrict__ ampblock_b,
                    const float* __restrict__ linamp2_W,
                    const float* __restrict__ linamp2_b,
                    const float* __restrict__ lin0_W,
                    const float* __restrict__ deform_W,
                    const float* __restrict__ deform_b,
                    const float* __restrict__ lin1a_W,
                    const float* __restrict__ lin1b_W,
                    const float* __restrict__ k2, const int* __restrict__ dflag,
                    float* __restrict__ out) {
  alignas(32) __shared__ _Float16 wT[NTILES * 512];
  __shared__ float act[8][512];   // per-wave [16 rows x 32 ch] f32 activations
  __shared__ float stg[8][64];    // per-wave staging
  __shared__ float sk2[KSZ * KSZ];

  const int tid = threadIdx.x;

  // ---- build weight tiles in WMMA B layout (f16) ----
  for (int e = tid; e < NTILES * 512; e += 256) {
    const int t  = e >> 9;
    const int rr = e & 511;
    const int ln = rr >> 4;                    // lane 0..31
    const int j  = rr & 15;
    const int k  = ((j >= 8) ? 8 : 0) + ((ln >= 16) ? 8 : 0) + j;  // K row 0..31
    const int n  = ln & 15;                    // N col within half
    float val = 0.f;
    if (t < 2)        { if (k < 4)  val = linamp1_W[(t * 16 + n) * 4 + k]; }
    else if (t < 8)   { const int i = (t - 2) >> 1, h = (t - 2) & 1;
                        val = ampblock_W[i * NNEU * NNEU + (h * 16 + n) * NNEU + k]; }
    else if (t == 8)  { if (n == 0) val = linamp2_W[k]; }
    else if (t < 11)  { const int h = t - 9;
                        if (k < 10) val = lin0_W[(h * 16 + n) * 10 + k]; }
    else if (t < 17)  { const int i = (t - 11) >> 1, h = (t - 11) & 1;
                        val = deform_W[i * NNEU * NNEU + (h * 16 + n) * NNEU + k]; }
    else              { if (n < 3)  val = lin1a_W[n * NNEU + k]; }
    wT[e] = (_Float16)val;
  }
  if (tid < KSZ * KSZ) sk2[tid] = k2[tid];
  __syncthreads();

  const int wave = tid >> 5;
  const int lane = tid & 31;
  const int tileId = blockIdx.x * 8 + wave;    // 16 rows per wave
  float* a_act = act[wave];
  float* a_stg = stg[wave];

  const int row = tileId * 16 + (lane & 15);   // b*NPTS + p
  const int b   = row / NPTS;
  const int p   = row - b * NPTS;

  float px0 = 0.f, px1 = 0.f, px2 = 0.f;
  if (lane < 16) {
    px0 = pos[p * 3 + 0]; px1 = pos[p * 3 + 1]; px2 = pos[p * 3 + 2];
  }

  // ================= amplitude head =================
  if (lane < 16) {
    float* dst = a_act + lane * 32;
    dst[0] = px0; dst[1] = px1; dst[2] = px2;
    dst[3] = z[b * LATD + (LATD - 1)];          // conf_a = z[:, -1]
#pragma unroll
    for (int kk = 4; kk < 32; ++kk) dst[kk] = 0.f;
  }
  v8f dAmp = run_head(a_act, wT, lane, /*tIn=*/0, /*tLay=*/2, ampblock_b, /*tOut=*/8);

  // extract column N=0 (lanes 0 and 16 own it) -> stg[0..15]
  if ((lane & 15) == 0) {
    const int rb = (lane >= 16) ? 8 : 0;
#pragma unroll
    for (int v = 0; v < 8; ++v) a_stg[rb + v] = dAmp[v];
  }
  float ampc = 0.f;
  if (lane < 16) {
    const float logit = a_stg[lane] + linamp2_b[0];
    ampc = 1.f / (1.f + expf(-logit));          // sigmoid
  }

  // ================= deformation head =================
  if (lane < 16) {
    float* dst = a_act + lane * 32;
    dst[0] = px0; dst[1] = px1; dst[2] = px2;
#pragma unroll
    for (int c = 0; c < LATD - 1; ++c) dst[3 + c] = z[b * LATD + c]; // z[:, :-1]
#pragma unroll
    for (int kk = 3 + LATD - 1; kk < 32; ++kk) dst[kk] = 0.f;
  }
  v8f dDef = run_head(a_act, wT, lane, /*tIn=*/9, /*tLay=*/11, deform_b, /*tOut=*/17);

  // extract columns N=0..2 (pre-tanh) -> stg[16 + row*3 + col]
  if ((lane & 15) < 3) {
    const int rb = (lane >= 16) ? 8 : 0;
    const int cc = lane & 15;
#pragma unroll
    for (int v = 0; v < 8; ++v) a_stg[16 + (rb + v) * 3 + cc] = dDef[v];
  }

  // ================= per-row finishing =================
  const float dmul = (dflag[0] > 0) ? 1.f : 0.f;
  if (lane < 16) {
    const float t0 = tanhf(a_stg[16 + lane * 3 + 0]);
    const float t1 = tanhf(a_stg[16 + lane * 3 + 1]);
    const float t2 = tanhf(a_stg[16 + lane * 3 + 2]);
    float resv[3], pd[3];
#pragma unroll
    for (int i = 0; i < 3; ++i) {
      resv[i] = dmul * (lin1b_W[i * 3 + 0] * t0 + lin1b_W[i * 3 + 1] * t1 +
                        lin1b_W[i * 3 + 2] * t2);
    }
    pd[0] = px0 + resv[0]; pd[1] = px1 + resv[1]; pd[2] = px2 + resv[2];
    const float prx = r[b * 9 + 0] * pd[0] + r[b * 9 + 1] * pd[1] + r[b * 9 + 2] * pd[2];
    const float pry = r[b * 9 + 3] * pd[0] + r[b * 9 + 4] * pd[1] + r[b * 9 + 5] * pd[2];
#pragma unroll
    for (int i = 0; i < 3; ++i) {
      out[OUT_PD_OFF  + row * 3 + i] = pd[i];
      out[OUT_RES_OFF + row * 3 + i] = resv[i];
    }
    out[OUT_AC_OFF + row] = ampc;
    // stage projection + amplitude for the splat phase (all 32 lanes help)
    a_stg[lane]      = prx;
    a_stg[16 + lane] = pry;
    a_stg[32 + lane] = amp[0] * ampc;
  }

  // ================= fused splat (9x9 Gaussian, atomic adds) =================
  {
    const int rsel = lane & 15;
    const float sx = a_stg[rsel];
    const float sy = a_stg[16 + rsel];
    const float sa = a_stg[32 + rsel];
    const int rowg = tileId * 16 + rsel;
    const int bb   = rowg / NPTS;
    const int cx = __float2int_rn((sx + 0.5f) * (float)(BOXN - 1)); // round-half-even
    const int cy = __float2int_rn((sy + 0.5f) * (float)(BOXN - 1));
    const int base = (lane < 16) ? 0 : 41;      // split 81 taps across lane pairs
    float* img = out + OUT_IMG_OFF + bb * BOXN * BOXN;
    for (int i = 0; i < 41; ++i) {
      const int tap = base + i;
      if (tap > 80) break;
      const int ox = tap % 9 - 4;
      const int oy = tap / 9 - 4;
      const int x = cx + ox, y = cy + oy;
      if (x >= 0 && x < BOXN && y >= 0 && y < BOXN)
        atomicAdd(&img[y * BOXN + x], sa * sk2[tap]);
    }
  }
}

extern "C" void kernel_launch(void* const* d_in, const int* in_sizes, int n_in,
                              void* d_out, int out_size, void* d_ws, size_t ws_size,
                              hipStream_t stream) {
  (void)in_sizes; (void)n_in; (void)out_size; (void)d_ws; (void)ws_size;
  const float* z          = (const float*)d_in[0];
  const float* r          = (const float*)d_in[1];
  const float* pos        = (const float*)d_in[2];
  const float* amp        = (const float*)d_in[3];
  const float* linamp1_W  = (const float*)d_in[4];
  const float* ampblock_W = (const float*)d_in[5];
  const float* ampblock_b = (const float*)d_in[6];
  const float* linamp2_W  = (const float*)d_in[7];
  const float* linamp2_b  = (const float*)d_in[8];
  const float* lin0_W     = (const float*)d_in[9];
  const float* deform_W   = (const float*)d_in[10];
  const float* deform_b   = (const float*)d_in[11];
  const float* lin1a_W    = (const float*)d_in[12];
  const float* lin1b_W    = (const float*)d_in[13];
  const float* k2         = (const float*)d_in[14];
  const int*   dflag      = (const int*)d_in[15];
  float* out = (float*)d_out;

  // 1) zero the [16,256,256] splat image region (atomically accumulated later)
  zero_img_kernel<<<(NBATCH * BOXN * BOXN) / 256, 256, 0, stream>>>(out + OUT_IMG_OFF);

  // 2) MLP heads on WMMA + projection + fused Gaussian splat
  //    320,000 rows total, 16 rows per wave, 8 waves per block -> 2500 blocks
  decoder_kernel<<<(NBATCH * NPTS) / 128, 256, 0, stream>>>(
      z, r, pos, amp, linamp1_W, ampblock_W, ampblock_b, linamp2_W, linamp2_b,
      lin0_W, deform_W, deform_b, lin1a_W, lin1b_W, k2, dflag, out);
}
